// SelfAttnHuggingFace_18433999634871
// MI455X (gfx1250) — compile-verified, matrix operand order checked by reading, not run on hardware
//
#include <hip/hip_runtime.h>
#include <hip/hip_bf16.h>
#include <hip/hip_fp16.h>

// ---------------------------------------------------------------------------
// SAGAN self-attention forward for MI455X (gfx1250), wave32 + WMMA f16/f32,
// async global->LDS (ASYNCcnt) staging for ALL GEMM tile loads.
//
// Pipeline:
//   K0  sigma power iteration -> inv-sigma scalars (ws)
//   K1  scale+convert weights to f16 (Wcat = [theta;phi;g] 384x512, Woh 512x256)
//   K1b x (fp32, [c][w]) -> x^T (f16, [w][c]) via LDS tile transpose
//   K2  Yconv[b] (384x2048 f16) = Wcat x X[b]   (WMMA, K=512; A,B async)
//   K3  maxpool2x2+reshape: phi_p^T (512s x 64k), g_p (256x512) f16
//   K4  logits (16t x 512s, K=64) -> softmax -> attn f16 (fused WMMA; B async)
//   K5  ag^T[b] (2048t x 256c) = (g_p x attn^T)^T  (WMMA; A,B async)
//   K6  out = x + gamma * (Woh x ag)           (WMMA, K=256; A,B async)
// ---------------------------------------------------------------------------

typedef __attribute__((ext_vector_type(16))) _Float16 v16h;
typedef __attribute__((ext_vector_type(8)))  _Float16 h8;
typedef __attribute__((ext_vector_type(8)))  float    v8f;

#define BATCH 32
#define CH    512
#define WW    2048
#define C8    64
#define C2    256
#define SS    512   // WW/4

__device__ __forceinline__ v8f vzero8() {
  v8f z;
#pragma unroll
  for (int i = 0; i < 8; ++i) z[i] = 0.0f;
  return z;
}

// Async 16-byte global -> LDS copy (per-lane addresses), tracked by ASYNCcnt.
// LDS offset = low 32 bits of the flat shared-memory address (LDS aperture).
__device__ __forceinline__ void async_g2l_b128(void* lds_ptr, const void* gptr) {
  unsigned ldsoff = (unsigned)(unsigned long long)lds_ptr;
  asm volatile("global_load_async_to_lds_b128 %0, %1, off"
               :: "v"(ldsoff), "v"(gptr) : "memory");
}
__device__ __forceinline__ void wait_async0() {
  asm volatile("s_wait_asynccnt 0x0" ::: "memory");
}

// Build a 16-half WMMA A/B fragment from an LDS row (k contiguous in the row).
// halves 0..7 -> k = base+0..7, halves 8..15 -> k = base+16..23,
// base = 8 if lane >= 16 else 0. Two ds_load_b128 per lane.
__device__ __forceinline__ v16h load_frag(const _Float16* rowbase, int lane) {
  const int base = ((lane >> 4) & 1) << 3;
  h8 lo = *(const h8*)(rowbase + base);
  h8 hi = *(const h8*)(rowbase + base + 16);
  v16h r;
#pragma unroll
  for (int i = 0; i < 8; ++i) { r[i] = lo[i]; r[i + 8] = hi[i]; }
  return r;
}

__device__ __forceinline__ v8f wmma16(v16h a, v16h b, v8f c) {
  return __builtin_amdgcn_wmma_f32_16x16x32_f16(
      /*neg_a=*/false, a, /*neg_b=*/false, b,
      /*c_mod=*/(short)0, c, /*reuse_a=*/false, /*reuse_b=*/false);
}

// ---------------------------------------------------------------------------
// K0: one spectral-norm power iteration per weight -> inv(sigma) scalars.
// ---------------------------------------------------------------------------
__global__ void sn_sigma_kernel(const float* wt, const float* wp,
                                const float* wg, const float* wo,
                                const float* ut, const float* up,
                                const float* ug, const float* uo,
                                float* sig) {
  const int which = blockIdx.x;
  const float* Wm; const float* u; int O, I;
  if      (which == 0) { Wm = wt; u = ut; O = 64;  I = 512; }
  else if (which == 1) { Wm = wp; u = up; O = 64;  I = 512; }
  else if (which == 2) { Wm = wg; u = ug; O = 256; I = 512; }
  else                 { Wm = wo; u = uo; O = 512; I = 256; }

  __shared__ float v[512];
  __shared__ float red[256];
  __shared__ float inv_tn;
  const int tid = threadIdx.x;

  for (int c = tid; c < I; c += 256) {           // t = W^T u
    float acc = 0.f;
    for (int o = 0; o < O; ++o) acc += Wm[o * I + c] * u[o];
    v[c] = acc;
  }
  __syncthreads();
  float loc = 0.f;
  for (int c = tid; c < I; c += 256) loc += v[c] * v[c];
  red[tid] = loc; __syncthreads();
  for (int s = 128; s > 0; s >>= 1) { if (tid < s) red[tid] += red[tid + s]; __syncthreads(); }
  if (tid == 0) inv_tn = 1.0f / (sqrtf(red[0]) + 1e-12f);
  __syncthreads();
  const float itn = inv_tn;

  float ssloc = 0.f;                             // s = W v ; sigma = (s.s)/(||s||+eps)
  for (int o = tid; o < O; o += 256) {
    float acc = 0.f;
    for (int c = 0; c < I; ++c) acc += Wm[o * I + c] * (v[c] * itn);
    ssloc += acc * acc;
  }
  red[tid] = ssloc; __syncthreads();
  for (int s = 128; s > 0; s >>= 1) { if (tid < s) red[tid] += red[tid + s]; __syncthreads(); }
  if (tid == 0) {
    float ss = red[0];
    float sigma = ss / (sqrtf(ss) + 1e-12f);
    sig[which] = 1.0f / sigma;
  }
}

// ---------------------------------------------------------------------------
// K1: fold 1/sigma into weights, convert to f16.
// ---------------------------------------------------------------------------
__global__ void wconv_kernel(const float* wt, const float* wp,
                             const float* wg, const float* wo,
                             const float* sig, _Float16* wcat, _Float16* woh) {
  const int i = blockIdx.x * 256 + threadIdx.x;
  if      (i < 32768)  wcat[i] = (_Float16)(wt[i] * sig[0]);
  else if (i < 65536)  wcat[i] = (_Float16)(wp[i - 32768] * sig[1]);
  else if (i < 196608) wcat[i] = (_Float16)(wg[i - 65536] * sig[2]);
  else if (i < 327680) woh[i - 196608] = (_Float16)(wo[i - 196608] * sig[3]);
}

// ---------------------------------------------------------------------------
// K1b: x (fp32, [c][w]) -> x^T (f16, [w][c]) with an LDS tile so both the
//      global read (along w) and the global write (along c) are coalesced.
// ---------------------------------------------------------------------------
__global__ void xcvt_kernel(const float* __restrict__ x, _Float16* __restrict__ xt) {
  const int b = blockIdx.z;
  const int c0 = blockIdx.y * 64;
  const int w0 = blockIdx.x * 32;
  __shared__ _Float16 tile[64 * 32];
  const int tid = threadIdx.x;
  const float* xb = x + (size_t)b * CH * WW;
  for (int e = tid; e < 2048; e += 256) {        // read: contiguous along w
    int c = e >> 5, w = e & 31;
    tile[c * 32 + w] = (_Float16)xb[(size_t)(c0 + c) * WW + w0 + w];
  }
  __syncthreads();
  _Float16* xtb = xt + (size_t)b * WW * CH;
  for (int e = tid; e < 2048; e += 256) {        // write: contiguous along c
    int w = e >> 6, c = e & 63;
    xtb[(size_t)(w0 + w) * CH + c0 + c] = tile[c * 32 + w];
  }
}

// ---------------------------------------------------------------------------
// K2: Yconv[b] = Wcat (384x512 f16) x X[b]; A and B staged via async b128.
// ---------------------------------------------------------------------------
__global__ void conv_gemm_kernel(const _Float16* __restrict__ wcat,
                                 const _Float16* __restrict__ xt,
                                 _Float16* __restrict__ yconv) {
  const int b = blockIdx.z;
  const int m0 = blockIdx.y * 64;
  const int n0 = blockIdx.x * 128;
  const int tid = threadIdx.x, wave = tid >> 5, lane = tid & 31;
  __shared__ alignas(16) _Float16 lds_a[64 * 32];
  __shared__ alignas(16) _Float16 lds_b[128 * 32];
  const _Float16* xtb = xt + (size_t)b * WW * CH;       // [w][c]

  v8f acc[4] = {vzero8(), vzero8(), vzero8(), vzero8()};

  for (int k0 = 0; k0 < 512; k0 += 32) {
    {                                              // A: 64x32 = 256 b128 chunks
      int m = tid >> 2, c = tid & 3;
      async_g2l_b128(&lds_a[m * 32 + c * 8],
                     wcat + (size_t)(m0 + m) * 512 + k0 + c * 8);
    }
#pragma unroll
    for (int it = 0; it < 2; ++it) {               // B: 128n x 32k = 512 chunks
      int idx = tid + it * 256;
      int n = idx >> 2, c = idx & 3;
      async_g2l_b128(&lds_b[n * 32 + c * 8],
                     xtb + (size_t)(n0 + n) * CH + k0 + c * 8);
    }
    wait_async0();
    __syncthreads();
    v16h bf = load_frag(lds_b + (wave * 16 + (lane & 15)) * 32, lane);
#pragma unroll
    for (int mt = 0; mt < 4; ++mt) {
      v16h af = load_frag(lds_a + (mt * 16 + (lane & 15)) * 32, lane);
      acc[mt] = wmma16(af, bf, acc[mt]);
    }
    __syncthreads();
  }

  _Float16* yb = yconv + (size_t)b * 384 * WW;
  const int n = n0 + wave * 16 + (lane & 15);
#pragma unroll
  for (int mt = 0; mt < 4; ++mt)
#pragma unroll
    for (int v = 0; v < 8; ++v) {
      int m = m0 + mt * 16 + ((lane >> 4) << 3) + v;
      yb[(size_t)m * WW + n] = (_Float16)acc[mt][v];
    }
}

// ---------------------------------------------------------------------------
// K3: maxpool(2,2)+reshape. phi written TRANSPOSED: php_t[b][s*64 + k].
//     g written row-major: ghp[b][c*512 + s].
// ---------------------------------------------------------------------------
__global__ void pool_kernel(const _Float16* __restrict__ yconv,
                            _Float16* __restrict__ php_t,
                            _Float16* __restrict__ ghp) {
  const size_t i = (size_t)blockIdx.x * 256 + threadIdx.x;
  const int b = blockIdx.y;
  const _Float16* yb = yconv + (size_t)b * 384 * WW;
  if (i < 32768) {                                 // phi: (64,2048)->(32,1024)->(64,512)
    int c2 = (int)(i >> 10), w2 = (int)(i & 1023);
    const _Float16* base = yb + (size_t)(64 + 2 * c2) * WW + 2 * w2;
    float m = fmaxf(fmaxf((float)base[0], (float)base[1]),
                    fmaxf((float)base[WW], (float)base[WW + 1]));
    int flat = c2 * 1024 + w2;
    int k = flat >> 9, s = flat & 511;
    php_t[(size_t)b * SS * C8 + s * C8 + k] = (_Float16)m;
  } else if (i < 163840) {                         // g: (256,2048)->(128,1024)->(256,512)
    size_t j = i - 32768;
    int c2 = (int)(j >> 10), w2 = (int)(j & 1023);
    const _Float16* base = yb + (size_t)(128 + 2 * c2) * WW + 2 * w2;
    float m = fmaxf(fmaxf((float)base[0], (float)base[1]),
                    fmaxf((float)base[WW], (float)base[WW + 1]));
    int flat = c2 * 1024 + w2;
    ghp[(size_t)b * C2 * SS + (flat >> 9) * SS + (flat & 511)] = (_Float16)m;
  }
}

// ---------------------------------------------------------------------------
// K4: fused logits (16t x 512s, K=64) + softmax + attn write (f16).
//     B staged via async b128 from transposed phi; LDS reused for logits.
// ---------------------------------------------------------------------------
__global__ void attn_kernel(const _Float16* __restrict__ yconv,
                            const _Float16* __restrict__ php_t,
                            _Float16* __restrict__ attn) {
  const int b = blockIdx.y;
  const int t0 = blockIdx.x * 16;
  const int tid = threadIdx.x, wave = tid >> 5, lane = tid & 31;

  __shared__ alignas(16) _Float16 lds_a[16 * 64];       // theta^T tile, pitch 64
  __shared__ alignas(16) char smem_big[512 * 32 * 2];   // B chunk / logits (32 KB)
  __shared__ float red[256];
  __shared__ float rowmax[16];
  __shared__ float rowsum[16];

  _Float16* lds_b = (_Float16*)smem_big;
  float* logits = (float*)smem_big;

  const _Float16* yb  = yconv + (size_t)b * 384 * WW;   // theta = rows [0,64)
  const _Float16* pbt = php_t + (size_t)b * SS * C8;    // [s][k]

  // Stage A transposed (tiny): lds_a[t_local*64 + k] = theta[k, t0+t_local]
  for (int e = tid; e < 1024; e += 256) {
    int i = e & 15, k = e >> 4;
    lds_a[i * 64 + k] = yb[(size_t)k * WW + t0 + i];
  }

  v8f acc[4] = {vzero8(), vzero8(), vzero8(), vzero8()};
  for (int kc = 0; kc < 2; ++kc) {
    __syncthreads();
#pragma unroll
    for (int it = 0; it < 8; ++it) {                    // B: 512s x 32k, async copy
      int idx = tid + it * 256;                         // 2048 b128 chunks
      int s = idx >> 2, c = idx & 3;
      async_g2l_b128(&lds_b[s * 32 + c * 8],
                     pbt + (size_t)s * C8 + kc * 32 + c * 8);
    }
    wait_async0();
    __syncthreads();
    v16h af = load_frag(lds_a + (lane & 15) * 64 + kc * 32, lane);
#pragma unroll
    for (int j = 0; j < 4; ++j) {
      int st = wave * 4 + j;
      v16h bf = load_frag(lds_b + (st * 16 + (lane & 15)) * 32, lane);
      acc[j] = wmma16(af, bf, acc[j]);
    }
  }
  __syncthreads();                                      // all reads of lds_b done

#pragma unroll
  for (int j = 0; j < 4; ++j) {                         // spill logits (fp32)
    int s = (wave * 4 + j) * 16 + (lane & 15);
#pragma unroll
    for (int v = 0; v < 8; ++v) {
      int m = ((lane >> 4) << 3) + v;
      logits[m * 512 + s] = acc[j][v];
    }
  }
  __syncthreads();

  const int r = tid >> 4, cs = tid & 15;                // softmax, 16 thr/row
  float lmax = -3.0e38f;
  for (int j = 0; j < 32; ++j) lmax = fmaxf(lmax, logits[r * 512 + cs + 16 * j]);
  red[r * 16 + cs] = lmax; __syncthreads();
  if (cs == 0) {
    float m = red[r * 16];
    for (int i = 1; i < 16; ++i) m = fmaxf(m, red[r * 16 + i]);
    rowmax[r] = m;
  }
  __syncthreads();
  const float mx = rowmax[r];
  float lsum = 0.f;
  for (int j = 0; j < 32; ++j) {
    int idx = r * 512 + cs + 16 * j;
    float e = __expf(logits[idx] - mx);
    logits[idx] = e;
    lsum += e;
  }
  red[r * 16 + cs] = lsum; __syncthreads();
  if (cs == 0) {
    float s = 0.f;
    for (int i = 0; i < 16; ++i) s += red[r * 16 + i];
    rowsum[r] = s;
  }
  __syncthreads();
  const float inv = 1.0f / rowsum[r];
  _Float16* ab = attn + (size_t)b * WW * SS;
  for (int j = 0; j < 32; ++j) {
    int s = cs + 16 * j;
    ab[(size_t)(t0 + r) * SS + s] = (_Float16)(logits[r * 512 + s] * inv);
  }
}

// ---------------------------------------------------------------------------
// K5: ag^T[b] (2048 x 256) = (g_p (256x512) x attn^T)^T.
//     A and B staged via async b128; output written transposed so the next
//     GEMM's B staging is a pure copy too.
// ---------------------------------------------------------------------------
__global__ void attng_gemm_kernel(const _Float16* __restrict__ ghp,
                                  const _Float16* __restrict__ attn,
                                  _Float16* __restrict__ agt) {
  const int b = blockIdx.z;
  const int m0 = blockIdx.y * 64;
  const int n0 = blockIdx.x * 128;
  const int tid = threadIdx.x, wave = tid >> 5, lane = tid & 31;
  __shared__ alignas(16) _Float16 lds_a[64 * 32];
  __shared__ alignas(16) _Float16 lds_b[128 * 32];
  const _Float16* A  = ghp + (size_t)b * C2 * SS;
  const _Float16* At = attn + (size_t)b * WW * SS;      // attn[t][s]

  v8f acc[4] = {vzero8(), vzero8(), vzero8(), vzero8()};

  for (int k0 = 0; k0 < 512; k0 += 32) {
    {                                                   // A: 256 b128 chunks
      int m = tid >> 2, c = tid & 3;
      async_g2l_b128(&lds_a[m * 32 + c * 8],
                     A + (size_t)(m0 + m) * SS + k0 + c * 8);
    }
#pragma unroll
    for (int it = 0; it < 2; ++it) {                    // B: 512 b128 chunks
      int idx = tid + it * 256;
      int n = idx >> 2, c = idx & 3;
      async_g2l_b128(&lds_b[n * 32 + c * 8],
                     At + (size_t)(n0 + n) * SS + k0 + c * 8);
    }
    wait_async0();
    __syncthreads();
    v16h bf = load_frag(lds_b + (wave * 16 + (lane & 15)) * 32, lane);
#pragma unroll
    for (int mt = 0; mt < 4; ++mt) {
      v16h af = load_frag(lds_a + (mt * 16 + (lane & 15)) * 32, lane);
      acc[mt] = wmma16(af, bf, acc[mt]);
    }
    __syncthreads();
  }

  _Float16* obt = agt + (size_t)b * WW * C2;            // [t][c]
  const int n = n0 + wave * 16 + (lane & 15);
#pragma unroll
  for (int mt = 0; mt < 4; ++mt) {
    h8 hh;
#pragma unroll
    for (int v = 0; v < 8; ++v) hh[v] = (_Float16)acc[mt][v];
    *(h8*)(obt + (size_t)n * C2 + m0 + mt * 16 + ((lane >> 4) << 3)) = hh;
  }
}

// ---------------------------------------------------------------------------
// K6: out = x + gamma * (Woh (512x256) x ag (256x2048)), fp32 epilogue.
//     B read from transposed ag^T; A and B staged via async b128.
// ---------------------------------------------------------------------------
__global__ void out_gemm_kernel(const _Float16* __restrict__ woh,
                                const _Float16* __restrict__ agt,
                                const float* __restrict__ x,
                                const float* __restrict__ gamma,
                                float* __restrict__ out) {
  const int b = blockIdx.z;
  const int m0 = blockIdx.y * 64;
  const int n0 = blockIdx.x * 128;
  const int tid = threadIdx.x, wave = tid >> 5, lane = tid & 31;
  __shared__ alignas(16) _Float16 lds_a[64 * 32];
  __shared__ alignas(16) _Float16 lds_b[128 * 32];
  const _Float16* Bt = agt + (size_t)b * WW * C2;       // [t][c]

  v8f acc[4] = {vzero8(), vzero8(), vzero8(), vzero8()};

  for (int k0 = 0; k0 < 256; k0 += 32) {
    {                                                   // A: 256 b128 chunks
      int m = tid >> 2, c = tid & 3;
      async_g2l_b128(&lds_a[m * 32 + c * 8],
                     woh + (size_t)(m0 + m) * 256 + k0 + c * 8);
    }
#pragma unroll
    for (int it = 0; it < 2; ++it) {                    // B: 512 b128 chunks
      int idx = tid + it * 256;
      int n = idx >> 2, c = idx & 3;
      async_g2l_b128(&lds_b[n * 32 + c * 8],
                     Bt + (size_t)(n0 + n) * C2 + k0 + c * 8);
    }
    wait_async0();
    __syncthreads();
    v16h bf = load_frag(lds_b + (wave * 16 + (lane & 15)) * 32, lane);
#pragma unroll
    for (int mt = 0; mt < 4; ++mt) {
      v16h af = load_frag(lds_a + (mt * 16 + (lane & 15)) * 32, lane);
      acc[mt] = wmma16(af, bf, acc[mt]);
    }
    __syncthreads();
  }

  const float gm = gamma[0];
  const float* xb = x + (size_t)b * CH * WW;
  float* ob = out + (size_t)b * CH * WW;
  const int n = n0 + wave * 16 + (lane & 15);
#pragma unroll
  for (int mt = 0; mt < 4; ++mt)
#pragma unroll
    for (int v = 0; v < 8; ++v) {
      int m = m0 + mt * 16 + ((lane >> 4) << 3) + v;
      size_t idx = (size_t)m * WW + n;
      ob[idx] = xb[idx] + gm * acc[mt][v];
    }
}

// ---------------------------------------------------------------------------
extern "C" void kernel_launch(void* const* d_in, const int* in_sizes, int n_in,
                              void* d_out, int out_size, void* d_ws, size_t ws_size,
                              hipStream_t stream) {
  const float* x  = (const float*)d_in[0];
  const float* wt = (const float*)d_in[1];
  const float* wp = (const float*)d_in[2];
  const float* wg = (const float*)d_in[3];
  const float* wo = (const float*)d_in[4];
  const float* ut = (const float*)d_in[5];
  const float* up = (const float*)d_in[6];
  const float* ug = (const float*)d_in[7];
  const float* uo = (const float*)d_in[8];
  const float* gamma = (const float*)d_in[9];
  float* out = (float*)d_out;

  char* ws = (char*)d_ws;
  size_t off = 0;
  float* sig = (float*)(ws + off);             off += 256;
  _Float16* wcat  = (_Float16*)(ws + off);     off += (size_t)384 * 512 * 2;
  _Float16* woh   = (_Float16*)(ws + off);     off += (size_t)512 * 256 * 2;
  _Float16* xt    = (_Float16*)(ws + off);     off += (size_t)BATCH * WW * CH * 2;
  _Float16* yconv = (_Float16*)(ws + off);     off += (size_t)BATCH * 384 * WW * 2;
  _Float16* php_t = (_Float16*)(ws + off);     off += (size_t)BATCH * SS * C8 * 2;
  _Float16* ghp   = (_Float16*)(ws + off);     off += (size_t)BATCH * C2 * SS * 2;
  _Float16* attn  = (_Float16*)(ws + off);     off += (size_t)BATCH * WW * SS * 2;
  _Float16* agt   = (_Float16*)(ws + off);     off += (size_t)BATCH * WW * C2 * 2;

  sn_sigma_kernel<<<4, 256, 0, stream>>>(wt, wp, wg, wo, ut, up, ug, uo, sig);
  wconv_kernel<<<1280, 256, 0, stream>>>(wt, wp, wg, wo, sig, wcat, woh);
  xcvt_kernel<<<dim3(64, 8, BATCH), 256, 0, stream>>>(x, xt);
  conv_gemm_kernel<<<dim3(16, 6, BATCH), 256, 0, stream>>>(wcat, xt, yconv);
  pool_kernel<<<dim3(640, BATCH), 256, 0, stream>>>(yconv, php_t, ghp);
  attn_kernel<<<dim3(128, BATCH), 256, 0, stream>>>(yconv, php_t, attn);
  attng_gemm_kernel<<<dim3(16, 4, BATCH), 256, 0, stream>>>(ghp, attn, agt);
  out_gemm_kernel<<<dim3(16, 8, BATCH), 256, 0, stream>>>(woh, agt, x, gamma, out);
}
